// GNet_2559800508580
// MI455X (gfx1250) — matrix operand order repaired and checked by reading, not compiled, not run on hardware
//
#include <hip/hip_runtime.h>
#include <hip/hip_bf16.h>
#include <cstdint>

// ---------------------------------------------------------------------------
// GNet forward on gfx1250: every GEMM runs through v_wmma_f32_16x16x32_f16.
// Activations f16, accumulation f32. Weights pre-swizzled to B-fragment
// layout (one contiguous 32B load per lane). Dual-accumulator inner loops
// give two independent WMMA chains per wave. Edge-concat staging of the
// contiguous pair-feature columns uses global_load_async_to_lds_b128.
// ---------------------------------------------------------------------------

typedef __attribute__((ext_vector_type(16))) _Float16 v16h;
typedef __attribute__((ext_vector_type(8)))  float    v8f;

#define NDET 16384
#define NEDGE (16384 * 32)
#define DEV static __device__ __forceinline__

// --- WMMA fragment index maps (per CDNA5 ISA 7.12.2) -----------------------
// A 16x32 f16: lanes 0-15 hold row M=lane, K in {0..7, 16..23};
//              lanes 16-31 hold row M=lane-16, K in {8..15, 24..31}.
DEV int kmapA(int lane, int i) {
  int p = i >> 1;
  int khalf = (lane & 16) ? 8 : 0;
  int kb = (p < 4) ? (khalf + 2 * p) : (16 + khalf + 2 * (p - 4));
  return kb + (i & 1);
}
// B 32x16 f16: same K order as A: lanes 0-15 hold K in {0..7, 16..23} of col N=lane,
// lanes 16-31 hold K in {8..15, 24..31}.
DEV int kmapB(int lane, int i) { return (i < 8 ? 0 : 16) + ((lane & 16) ? 8 : 0) + (i & 7); }

DEV v8f wmma(v16h a, v16h b, v8f c) {
  return __builtin_amdgcn_wmma_f32_16x16x32_f16(
      /*neg_a=*/false, a, /*neg_b=*/false, b,
      /*c_mod=*/(short)0, c, /*reuse_a=*/false, /*reuse_b=*/false);
}

// A fragment from a row-major f16 tile (LDS or global), row stride lda elems.
DEV v16h load_a_h(const _Float16* base, int lda) {
  int lane = threadIdx.x & 31;
  const _Float16* row = base + (size_t)(lane & 15) * lda;
  int khalf = (lane & 16) ? 8 : 0;
  v16h a;
#pragma unroll
  for (int p = 0; p < 8; ++p) {
    int k = (p < 4) ? (khalf + 2 * p) : (16 + khalf + 2 * (p - 4));
    a[2 * p]     = row[k];
    a[2 * p + 1] = row[k + 1];
  }
  return a;
}

// A fragment straight from row-major f32 data, zero-padding K >= kvalid.
DEV v16h load_a_f32(const float* base, int lda, int kvalid) {
  int lane = threadIdx.x & 31;
  const float* row = base + (size_t)(lane & 15) * lda;
  v16h a;
#pragma unroll
  for (int i = 0; i < 16; ++i) {
    int k = kmapA(lane, i);
    a[i] = (_Float16)(k < kvalid ? row[k] : 0.0f);
  }
  return a;
}

// B fragment from pre-swizzled weights: 16 contiguous f16 per lane.
DEV v16h load_b_sw(const _Float16* frag) {
  return ((const v16h*)frag)[threadIdx.x & 31];
}

// C store (bias + relu) to a row-major f16 tile; bv = bias for this lane's N.
DEV void store_relu_h(_Float16* base, int ldc, v8f c, float bv) {
  int lane = threadIdx.x & 31;
  int n = lane & 15, m0 = (lane & 16) ? 8 : 0;
#pragma unroll
  for (int r = 0; r < 8; ++r) {
    float v = c[r] + bv;
    base[(size_t)(m0 + r) * ldc + n] = (_Float16)(v > 0.0f ? v : 0.0f);
  }
}

// C store with residual add + relu into f32 global (x update).
DEV void store_res_relu_f32(float* base, int ldc, v8f c, float bv) {
  int lane = threadIdx.x & 31;
  int n = lane & 15, m0 = (lane & 16) ? 8 : 0;
#pragma unroll
  for (int r = 0; r < 8; ++r) {
    size_t idx = (size_t)(m0 + r) * ldc + n;
    float v = base[idx] + c[r] + bv;
    base[idx] = v > 0.0f ? v : 0.0f;
  }
}

// --- weight pre-swizzle: f32 (K,N) row-major -> f16 B-fragment tiles -------
__global__ void k_swizzle(const float* __restrict__ W, _Float16* __restrict__ out,
                          int K, int N, int nnt) {
  int t = blockIdx.x * blockDim.x + threadIdx.x;
  int i = t & 15;
  int lane = (t >> 4) & 31;
  int tile = t >> 9;
  int nt = tile % nnt, kt = tile / nnt;
  int k = kt * 32 + kmapB(lane, i);
  int n = nt * 16 + (lane & 15);
  float v = (k < K && n < N) ? W[(size_t)k * N + n] : 0.0f;
  out[t] = (_Float16)v;
}

// --- fused pairwise-feature MLP: (E,9)->256->256->32 -----------------------
// 128 edges per workgroup; 8 waves, each owning one 16-row M-tile.
__global__ __launch_bounds__(256) void k_pair(
    const float* __restrict__ pairRaw,
    const _Float16* __restrict__ gW1s, const float* __restrict__ gb1,
    const _Float16* __restrict__ gW2s, const float* __restrict__ gb2,
    const _Float16* __restrict__ gW3s, const float* __restrict__ gb3,
    _Float16* __restrict__ pair16) {
  extern __shared__ _Float16 smem[];
  _Float16* act1 = smem;              // 128 x 256
  _Float16* act2 = smem + 128 * 256;  // 128 x 256
  const int w = threadIdx.x >> 5, lane = threadIdx.x & 31, n_l = lane & 15;
  const int row0 = w * 16;
  const size_t e0 = (size_t)blockIdx.x * 128;

  // L1: (16x9 pad32) @ (32x256), dual accumulators.
  {
    v16h a = load_a_f32(pairRaw + (e0 + row0) * 9, 9, 9);
#pragma unroll
    for (int nt = 0; nt < 16; nt += 2) {
      v8f acc0 = {}, acc1 = {};
      acc0 = wmma(a, load_b_sw(gW1s + (size_t)nt * 512), acc0);
      acc1 = wmma(a, load_b_sw(gW1s + (size_t)(nt + 1) * 512), acc1);
      store_relu_h(act1 + (size_t)row0 * 256 + nt * 16, 256, acc0, gb1[nt * 16 + n_l]);
      store_relu_h(act1 + (size_t)row0 * 256 + (nt + 1) * 16, 256, acc1,
                   gb1[(nt + 1) * 16 + n_l]);
    }
  }
  // L2: (16x256) @ (256x256)
  {
    v16h a8[8];
#pragma unroll
    for (int kt = 0; kt < 8; ++kt)
      a8[kt] = load_a_h(act1 + (size_t)row0 * 256 + kt * 32, 256);
    for (int nt = 0; nt < 16; nt += 2) {
      v8f acc0 = {}, acc1 = {};
#pragma unroll
      for (int kt = 0; kt < 8; ++kt) {
        acc0 = wmma(a8[kt], load_b_sw(gW2s + ((size_t)kt * 16 + nt) * 512), acc0);
        acc1 = wmma(a8[kt], load_b_sw(gW2s + ((size_t)kt * 16 + nt + 1) * 512), acc1);
      }
      store_relu_h(act2 + (size_t)row0 * 256 + nt * 16, 256, acc0, gb2[nt * 16 + n_l]);
      store_relu_h(act2 + (size_t)row0 * 256 + (nt + 1) * 16, 256, acc1,
                   gb2[(nt + 1) * 16 + n_l]);
    }
  }
  // L3: (16x256) @ (256x32) -> global f16
  {
    v16h a8[8];
#pragma unroll
    for (int kt = 0; kt < 8; ++kt)
      a8[kt] = load_a_h(act2 + (size_t)row0 * 256 + kt * 32, 256);
    v8f acc0 = {}, acc1 = {};
#pragma unroll
    for (int kt = 0; kt < 8; ++kt) {
      acc0 = wmma(a8[kt], load_b_sw(gW3s + (size_t)(kt * 2 + 0) * 512), acc0);
      acc1 = wmma(a8[kt], load_b_sw(gW3s + (size_t)(kt * 2 + 1) * 512), acc1);
    }
    store_relu_h(pair16 + (e0 + row0) * 32 + 0, 32, acc0, gb3[n_l]);
    store_relu_h(pair16 + (e0 + row0) * 32 + 16, 32, acc1, gb3[16 + n_l]);
  }
}

// --- per-block fc1: f = relu(x @ W1) : (N,128)@(128,32) -> f16 -------------
__global__ __launch_bounds__(256) void k_fc1(
    const float* __restrict__ x, const _Float16* __restrict__ Ws,
    const float* __restrict__ bias, _Float16* __restrict__ f) {
  const int w = threadIdx.x >> 5, lane = threadIdx.x & 31, n_l = lane & 15;
  const size_t r0 = (size_t)blockIdx.x * 128 + w * 16;
  v16h a4[4];
#pragma unroll
  for (int kt = 0; kt < 4; ++kt)
    a4[kt] = load_a_f32(x + r0 * 128 + kt * 32, 128, 32);
  v8f acc0 = {}, acc1 = {};
#pragma unroll
  for (int kt = 0; kt < 4; ++kt) {
    acc0 = wmma(a4[kt], load_b_sw(Ws + (size_t)(kt * 2 + 0) * 512), acc0);
    acc1 = wmma(a4[kt], load_b_sw(Ws + (size_t)(kt * 2 + 1) * 512), acc1);
  }
  store_relu_h(f + r0 * 32 + 0, 32, acc0, bias[n_l]);
  store_relu_h(f + r0 * 32 + 16, 32, acc1, bias[16 + n_l]);
}

// --- per-block edge MLP + segment-max --------------------------------------
// 128 edges / WG = exactly 4 detections (cIdxs = repeat(arange, 32)).
__global__ __launch_bounds__(256) void k_edge(
    const _Float16* __restrict__ pair16, const _Float16* __restrict__ f,
    const int* __restrict__ cIdxs, const int* __restrict__ nIdxs,
    const _Float16* __restrict__ W1s, const float* __restrict__ b1,
    const _Float16* __restrict__ W2s, const float* __restrict__ b2,
    _Float16* __restrict__ m16) {
  __shared__ _Float16 h0[128 * 96];
  __shared__ _Float16 h1[128 * 64];
  __shared__ _Float16 h2[128 * 64];
  const size_t e0 = (size_t)blockIdx.x * 128;
  // Build concat tile [pair | f[c] | (c==n ? 0 : f[n])].
  {
    int r = threadIdx.x >> 1, half = threadIdx.x & 1;
    size_t e = e0 + r;
    // pair columns (contiguous 64B/row): async DMA straight into LDS,
    // 16B per lane, tracked by ASYNCcnt.
    unsigned ldsoff = (unsigned)(uintptr_t)&h0[r * 96 + half * 16];
    unsigned long long gsrc =
        (unsigned long long)(uintptr_t)(pair16 + e * 32 + half * 16);
    asm volatile("global_load_async_to_lds_b128 %0, %1, off"
                 :: "v"(ldsoff), "v"(gsrc) : "memory");
    // gathered columns through the normal vector path
    int c = cIdxs[e], nn = nIdxs[e];
    bool self = (c == nn);
    const _Float16* src = half ? (f + (size_t)nn * 32) : (f + (size_t)c * 32);
    _Float16* dst = &h0[r * 96 + 32 + half * 32];
    if (half && self) {
      for (int j = 0; j < 32; ++j) dst[j] = (_Float16)0.0f;
    } else {
      for (int j = 0; j < 32; ++j) dst[j] = src[j];
    }
  }
  asm volatile("s_wait_asynccnt 0x0" ::: "memory");
  __syncthreads();
  const int w = threadIdx.x >> 5, lane = threadIdx.x & 31, n_l = lane & 15;
  const int row0 = w * 16;
  // L1: (16x96) @ (96x64)
  {
    v16h a3[3];
#pragma unroll
    for (int kt = 0; kt < 3; ++kt) a3[kt] = load_a_h(h0 + row0 * 96 + kt * 32, 96);
#pragma unroll
    for (int nt = 0; nt < 4; nt += 2) {
      v8f acc0 = {}, acc1 = {};
#pragma unroll
      for (int kt = 0; kt < 3; ++kt) {
        acc0 = wmma(a3[kt], load_b_sw(W1s + ((size_t)kt * 4 + nt) * 512), acc0);
        acc1 = wmma(a3[kt], load_b_sw(W1s + ((size_t)kt * 4 + nt + 1) * 512), acc1);
      }
      store_relu_h(h1 + row0 * 64 + nt * 16, 64, acc0, b1[nt * 16 + n_l]);
      store_relu_h(h1 + row0 * 64 + (nt + 1) * 16, 64, acc1, b1[(nt + 1) * 16 + n_l]);
    }
  }
  // L2: (16x64) @ (64x64)
  {
    v16h a2[2];
#pragma unroll
    for (int kt = 0; kt < 2; ++kt) a2[kt] = load_a_h(h1 + row0 * 64 + kt * 32, 64);
#pragma unroll
    for (int nt = 0; nt < 4; nt += 2) {
      v8f acc0 = {}, acc1 = {};
#pragma unroll
      for (int kt = 0; kt < 2; ++kt) {
        acc0 = wmma(a2[kt], load_b_sw(W2s + ((size_t)kt * 4 + nt) * 512), acc0);
        acc1 = wmma(a2[kt], load_b_sw(W2s + ((size_t)kt * 4 + nt + 1) * 512), acc1);
      }
      store_relu_h(h2 + row0 * 64 + nt * 16, 64, acc0, b2[nt * 16 + n_l]);
      store_relu_h(h2 + row0 * 64 + (nt + 1) * 16, 64, acc1, b2[(nt + 1) * 16 + n_l]);
    }
  }
  __syncthreads();
  // Segment max over contiguous 32-edge groups -> (4, 64) per WG.
  {
    int g = threadIdx.x >> 6, j = threadIdx.x & 63;
    float mx = -3.0e38f;
#pragma unroll 4
    for (int r = 0; r < 32; ++r) {
      float v = (float)h2[(g * 32 + r) * 64 + j];
      mx = v > mx ? v : mx;
    }
    m16[((size_t)blockIdx.x * 4 + g) * 64 + j] = (_Float16)mx;
  }
}

// --- per-block post MLP + residual update of x -----------------------------
__global__ __launch_bounds__(256) void k_post(
    const _Float16* __restrict__ m16,
    const _Float16* __restrict__ W1s, const float* __restrict__ b1,
    const _Float16* __restrict__ W2s, const float* __restrict__ b2,
    const _Float16* __restrict__ Wos, const float* __restrict__ bo,
    float* __restrict__ x) {
  __shared__ _Float16 p1[128 * 64];
  __shared__ _Float16 p2[128 * 64];
  const int w = threadIdx.x >> 5, lane = threadIdx.x & 31, n_l = lane & 15;
  const int row0 = w * 16;
  const size_t r0 = (size_t)blockIdx.x * 128 + row0;
  {
    v16h a2[2];
#pragma unroll
    for (int kt = 0; kt < 2; ++kt) a2[kt] = load_a_h(m16 + r0 * 64 + kt * 32, 64);
#pragma unroll
    for (int nt = 0; nt < 4; nt += 2) {
      v8f acc0 = {}, acc1 = {};
#pragma unroll
      for (int kt = 0; kt < 2; ++kt) {
        acc0 = wmma(a2[kt], load_b_sw(W1s + ((size_t)kt * 4 + nt) * 512), acc0);
        acc1 = wmma(a2[kt], load_b_sw(W1s + ((size_t)kt * 4 + nt + 1) * 512), acc1);
      }
      store_relu_h(p1 + row0 * 64 + nt * 16, 64, acc0, b1[nt * 16 + n_l]);
      store_relu_h(p1 + row0 * 64 + (nt + 1) * 16, 64, acc1, b1[(nt + 1) * 16 + n_l]);
    }
  }
  {
    v16h a2[2];
#pragma unroll
    for (int kt = 0; kt < 2; ++kt) a2[kt] = load_a_h(p1 + row0 * 64 + kt * 32, 64);
#pragma unroll
    for (int nt = 0; nt < 4; nt += 2) {
      v8f acc0 = {}, acc1 = {};
#pragma unroll
      for (int kt = 0; kt < 2; ++kt) {
        acc0 = wmma(a2[kt], load_b_sw(W2s + ((size_t)kt * 4 + nt) * 512), acc0);
        acc1 = wmma(a2[kt], load_b_sw(W2s + ((size_t)kt * 4 + nt + 1) * 512), acc1);
      }
      store_relu_h(p2 + row0 * 64 + nt * 16, 64, acc0, b2[nt * 16 + n_l]);
      store_relu_h(p2 + row0 * 64 + (nt + 1) * 16, 64, acc1, b2[(nt + 1) * 16 + n_l]);
    }
  }
  {
    v16h a2[2];
#pragma unroll
    for (int kt = 0; kt < 2; ++kt) a2[kt] = load_a_h(p2 + row0 * 64 + kt * 32, 64);
#pragma unroll
    for (int nt = 0; nt < 8; nt += 2) {
      v8f acc0 = {}, acc1 = {};
#pragma unroll
      for (int kt = 0; kt < 2; ++kt) {
        acc0 = wmma(a2[kt], load_b_sw(Wos + ((size_t)kt * 8 + nt) * 512), acc0);
        acc1 = wmma(a2[kt], load_b_sw(Wos + ((size_t)kt * 8 + nt + 1) * 512), acc1);
      }
      store_res_relu_f32(x + r0 * 128 + nt * 16, 128, acc0, bo[nt * 16 + n_l]);
      store_res_relu_f32(x + r0 * 128 + (nt + 1) * 16, 128, acc1, bo[(nt + 1) * 16 + n_l]);
    }
  }
}

// --- score head: 3x (128->128 relu) + 128->1 (WMMA vs zero-padded pred_W) --
__global__ __launch_bounds__(256) void k_score(
    const float* __restrict__ x,
    const _Float16* __restrict__ sWs, const float* __restrict__ sb,
    const _Float16* __restrict__ pWs, const float* __restrict__ pb,
    float* __restrict__ out) {
  extern __shared__ _Float16 smem[];
  _Float16* bufA = smem;              // 128 x 128
  _Float16* bufB = smem + 128 * 128;  // 128 x 128
  const int w = threadIdx.x >> 5, lane = threadIdx.x & 31, n_l = lane & 15;
  const int row0 = w * 16;
  const size_t r0 = (size_t)blockIdx.x * 128 + row0;
  // layer 0: A from f32 x
  {
    v16h a4[4];
#pragma unroll
    for (int kt = 0; kt < 4; ++kt) a4[kt] = load_a_f32(x + r0 * 128 + kt * 32, 128, 32);
#pragma unroll
    for (int nt = 0; nt < 8; nt += 2) {
      v8f acc0 = {}, acc1 = {};
#pragma unroll
      for (int kt = 0; kt < 4; ++kt) {
        acc0 = wmma(a4[kt], load_b_sw(sWs + ((size_t)kt * 8 + nt) * 512), acc0);
        acc1 = wmma(a4[kt], load_b_sw(sWs + ((size_t)kt * 8 + nt + 1) * 512), acc1);
      }
      store_relu_h(bufA + row0 * 128 + nt * 16, 128, acc0, sb[nt * 16 + n_l]);
      store_relu_h(bufA + row0 * 128 + (nt + 1) * 16, 128, acc1, sb[(nt + 1) * 16 + n_l]);
    }
  }
  _Float16* cur = bufA;
  _Float16* nxt = bufB;
  for (int l = 1; l < 3; ++l) {
    const _Float16* Wl = sWs + (size_t)l * 16384;
    const float* bl = sb + l * 128;
    v16h a4[4];
#pragma unroll
    for (int kt = 0; kt < 4; ++kt) a4[kt] = load_a_h(cur + row0 * 128 + kt * 32, 128);
#pragma unroll
    for (int nt = 0; nt < 8; nt += 2) {
      v8f acc0 = {}, acc1 = {};
#pragma unroll
      for (int kt = 0; kt < 4; ++kt) {
        acc0 = wmma(a4[kt], load_b_sw(Wl + ((size_t)kt * 8 + nt) * 512), acc0);
        acc1 = wmma(a4[kt], load_b_sw(Wl + ((size_t)kt * 8 + nt + 1) * 512), acc1);
      }
      store_relu_h(nxt + row0 * 128 + nt * 16, 128, acc0, bl[nt * 16 + n_l]);
      store_relu_h(nxt + row0 * 128 + (nt + 1) * 16, 128, acc1, bl[(nt + 1) * 16 + n_l]);
    }
    _Float16* t = cur; cur = nxt; nxt = t;
  }
  // pred: (16x128) @ (128x16) where only column 0 is real
  {
    v16h a4[4];
#pragma unroll
    for (int kt = 0; kt < 4; ++kt) a4[kt] = load_a_h(cur + row0 * 128 + kt * 32, 128);
    v8f acc = {};
#pragma unroll
    for (int kt = 0; kt < 4; ++kt)
      acc = wmma(a4[kt], load_b_sw(pWs + (size_t)kt * 512), acc);
    if (n_l == 0) {
      int m0 = (lane & 16) ? 8 : 0;
      float b0 = pb[0];
#pragma unroll
      for (int r = 0; r < 8; ++r) out[r0 + m0 + r] = acc[r] + b0;
    }
  }
}

// ---------------------------------------------------------------------------
extern "C" void kernel_launch(void* const* d_in, const int* in_sizes, int n_in,
                              void* d_out, int out_size, void* d_ws, size_t ws_size,
                              hipStream_t stream) {
  (void)in_sizes; (void)n_in; (void)out_size; (void)ws_size;
  const float* detFeatures = (const float*)d_in[0];
  const float* pairRaw     = (const float*)d_in[1];
  const int*   cIdxs       = (const int*)d_in[2];
  const int*   nIdxs       = (const int*)d_in[3];
  const float* g_W1 = (const float*)d_in[4];  const float* g_b1 = (const float*)d_in[5];
  const float* g_W2 = (const float*)d_in[6];  const float* g_b2 = (const float*)d_in[7];
  const float* g_W3 = (const float*)d_in[8];  const float* g_b3 = (const float*)d_in[9];
  const float* blk_fc1_W  = (const float*)d_in[10]; const float* blk_fc1_b  = (const float*)d_in[11];
  const float* blk_pw_W1  = (const float*)d_in[12]; const float* blk_pw_b1  = (const float*)d_in[13];
  const float* blk_pw_W2  = (const float*)d_in[14]; const float* blk_pw_b2  = (const float*)d_in[15];
  const float* blk_post_W1 = (const float*)d_in[16]; const float* blk_post_b1 = (const float*)d_in[17];
  const float* blk_post_W2 = (const float*)d_in[18]; const float* blk_post_b2 = (const float*)d_in[19];
  const float* blk_out_W  = (const float*)d_in[20]; const float* blk_out_b  = (const float*)d_in[21];
  const float* s_W = (const float*)d_in[22]; const float* s_b = (const float*)d_in[23];
  const float* pred_W = (const float*)d_in[24]; const float* pred_b = (const float*)d_in[25];

  char* ws = (char*)d_ws;
  size_t off = 0;
  auto alloc = [&](size_t bytes) -> void* {
    void* p = ws + off;
    off = (off + bytes + 255) & ~(size_t)255;
    return p;
  };
  _Float16* gW1s   = (_Float16*)alloc(16 * 512 * 2);           // 32x256
  _Float16* gW2s   = (_Float16*)alloc(128 * 512 * 2);          // 256x256
  _Float16* gW3s   = (_Float16*)alloc(16 * 512 * 2);           // 256x32
  _Float16* fc1s   = (_Float16*)alloc(4 * 8 * 512 * 2);        // 4x 128x32
  _Float16* pw1s   = (_Float16*)alloc(4 * 12 * 512 * 2);       // 4x 96x64
  _Float16* pw2s   = (_Float16*)alloc(4 * 8 * 512 * 2);        // 4x 64x64
  _Float16* post1s = (_Float16*)alloc(4 * 8 * 512 * 2);
  _Float16* post2s = (_Float16*)alloc(4 * 8 * 512 * 2);
  _Float16* outs   = (_Float16*)alloc(4 * 16 * 512 * 2);       // 4x 64x128
  _Float16* sWs    = (_Float16*)alloc(3 * 32 * 512 * 2);       // 3x 128x128
  _Float16* predWs = (_Float16*)alloc(4 * 512 * 2);            // 128x16 (N padded)
  _Float16* pair16 = (_Float16*)alloc((size_t)NEDGE * 32 * 2); // 32 MB
  _Float16* f16f   = (_Float16*)alloc((size_t)NDET * 32 * 2);
  _Float16* m16    = (_Float16*)alloc((size_t)NDET * 64 * 2);
  float*    xbuf   = (float*)alloc((size_t)NDET * 128 * 4);

  auto swz = [&](const float* W, _Float16* o, int K, int N, int Kpad, int Npad) {
    int tiles = (Kpad / 32) * (Npad / 16);
    k_swizzle<<<dim3(tiles * 2), dim3(256), 0, stream>>>(W, o, K, N, Npad / 16);
  };
  swz(g_W1, gW1s, 9, 256, 32, 256);
  swz(g_W2, gW2s, 256, 256, 256, 256);
  swz(g_W3, gW3s, 256, 32, 256, 32);
  for (int b = 0; b < 4; ++b) {
    swz(blk_fc1_W  + (size_t)b * 128 * 32, fc1s   + (size_t)b * 8 * 512, 128, 32, 128, 32);
    swz(blk_pw_W1  + (size_t)b * 96 * 64,  pw1s   + (size_t)b * 12 * 512, 96, 64, 96, 64);
    swz(blk_pw_W2  + (size_t)b * 64 * 64,  pw2s   + (size_t)b * 8 * 512,  64, 64, 64, 64);
    swz(blk_post_W1 + (size_t)b * 64 * 64, post1s + (size_t)b * 8 * 512,  64, 64, 64, 64);
    swz(blk_post_W2 + (size_t)b * 64 * 64, post2s + (size_t)b * 8 * 512,  64, 64, 64, 64);
    swz(blk_out_W  + (size_t)b * 64 * 128, outs   + (size_t)b * 16 * 512, 64, 128, 64, 128);
  }
  for (int l = 0; l < 3; ++l)
    swz(s_W + (size_t)l * 128 * 128, sWs + (size_t)l * 32 * 512, 128, 128, 128, 128);
  swz(pred_W, predWs, 128, 1, 128, 16);

  hipMemcpyAsync(xbuf, detFeatures, (size_t)NDET * 128 * 4,
                 hipMemcpyDeviceToDevice, stream);

  k_pair<<<dim3(NEDGE / 128), dim3(256), 2 * 128 * 256 * 2, stream>>>(
      pairRaw, gW1s, g_b1, gW2s, g_b2, gW3s, g_b3, pair16);

  for (int b = 0; b < 4; ++b) {
    k_fc1<<<dim3(NDET / 128), dim3(256), 0, stream>>>(
        xbuf, fc1s + (size_t)b * 8 * 512, blk_fc1_b + (size_t)b * 32, f16f);
    k_edge<<<dim3(NEDGE / 128), dim3(256), 0, stream>>>(
        pair16, f16f, cIdxs, nIdxs,
        pw1s + (size_t)b * 12 * 512, blk_pw_b1 + (size_t)b * 64,
        pw2s + (size_t)b * 8 * 512,  blk_pw_b2 + (size_t)b * 64, m16);
    k_post<<<dim3(NDET / 128), dim3(256), 0, stream>>>(
        m16,
        post1s + (size_t)b * 8 * 512, blk_post_b1 + (size_t)b * 64,
        post2s + (size_t)b * 8 * 512, blk_post_b2 + (size_t)b * 64,
        outs + (size_t)b * 16 * 512,  blk_out_b + (size_t)b * 128, xbuf);
  }

  k_score<<<dim3(NDET / 128), dim3(256), 2 * 128 * 128 * 2, stream>>>(
      xbuf, sWs, s_b, predWs, pred_b, (float*)d_out);
}
